// SeqAttention_56813827392291
// MI455X (gfx1250) — matrix operand order repaired and checked by reading, not compiled, not run on hardware
//
#include <hip/hip_runtime.h>
#include <hip/hip_bf16.h>

typedef __attribute__((ext_vector_type(16))) _Float16 v16h;
typedef __attribute__((ext_vector_type(8)))  _Float16 v8h;
typedef __attribute__((ext_vector_type(8)))  float    v8f;

#define C_S 256
#define C_Z 128
#define HEADS 8
#define DH 32
#define S_DIM 128
#define R_DIM 256
#define NTOK (S_DIM * R_DIM)   // 32768
#define INFV 1000000000.0f

// ---------------------------------------------------------------------------
// WMMA fragment loaders matching CDNA5 16-bit layouts (CDNA5 ISA)
// A 16x32 (MxK): lanes 0-15 -> M=lane, halves 0-7 = K[k..k+7], 8-15 = K[k+16..k+23]
//                lanes16-31 -> M=lane-16, K offsets +8 / +24
// rowPtr must already point at A[row_of_this_lane][0]; row stride = ldk halves.
__device__ inline v16h load_frag_A(const _Float16* rowPtr, int kBase, int lane) {
    int off0 = kBase + ((lane < 16) ? 0 : 8);
    v8h lo = *(const v8h*)(rowPtr + off0);
    v8h hi = *(const v8h*)(rowPtr + off0 + 16);
    return __builtin_shufflevector(lo, hi, 0,1,2,3,4,5,6,7,8,9,10,11,12,13,14,15);
}
// B 32x16 (KxN), stored as BT[n][k] (K contiguous per n-row).
// lane holds N = lane&15; halves 0-7 = K[k..k+7], 8-15 = K[k+16..k+23] (lanes16-31: +8 / +24).
// ntRowPtr points at BT[n_of_this_lane][0].
__device__ inline v16h load_frag_B(const _Float16* ntRowPtr, int kBase, int lane) {
    int off = kBase + ((lane < 16) ? 0 : 8);
    v8h lo = *(const v8h*)(ntRowPtr + off);
    v8h hi = *(const v8h*)(ntRowPtr + off + 16);
    return __builtin_shufflevector(lo, hi, 0,1,2,3,4,5,6,7,8,9,10,11,12,13,14,15);
}
__device__ inline v8f wmma_f16(v16h a, v16h b, v8f c) {
    return __builtin_amdgcn_wmma_f32_16x16x32_f16(false, a, false, b, (short)0, c, false, false);
}

// ---------------------------------------------------------------------------
// Kernel 1: transpose+cast weights to f16 BT layout [n][k]; fold 1/sqrt(D) into w_q.
// wT holds 5 matrices of 256x256 halves: q,k,v,g,o
__global__ void prep_w_kernel(const float* __restrict__ wq, const float* __restrict__ wk,
                              const float* __restrict__ wv, const float* __restrict__ wg,
                              const float* __restrict__ wo, _Float16* __restrict__ wT) {
    int tid = blockIdx.x * 256 + threadIdx.x;     // 5*65536 total
    int w   = tid >> 16;
    int rem = tid & 65535;
    int n = rem >> 8, k = rem & 255;
    const float* src = (w == 0) ? wq : (w == 1) ? wk : (w == 2) ? wv : (w == 3) ? wg : wo;
    float scale = (w == 0) ? 0.17677669529663687f : 1.0f;   // 1/sqrt(32) folded into q
    wT[tid] = (_Float16)(src[k * 256 + n] * scale);
}

// ---------------------------------------------------------------------------
// Kernel 2: layernorm over last dim of s (256), cast to f16. One block per row.
__global__ void ln_s_kernel(const float* __restrict__ s, const float* __restrict__ w,
                            const float* __restrict__ b, _Float16* __restrict__ out) {
    int row = blockIdx.x, t = threadIdx.x;
    float x = s[row * 256 + t];
    __shared__ float rs[256], rq[256];
    rs[t] = x; rq[t] = x * x;
    __syncthreads();
    for (int st = 128; st > 0; st >>= 1) {
        if (t < st) { rs[t] += rs[t + st]; rq[t] += rq[t + st]; }
        __syncthreads();
    }
    float mean = rs[0] * (1.0f / 256.0f);
    float var  = rq[0] * (1.0f / 256.0f) - mean * mean;
    float rstd = rsqrtf(var + 1e-5f);
    out[row * 256 + t] = (_Float16)((x - mean) * rstd * w[t] + b[t]);
}

// ---------------------------------------------------------------------------
// Kernel 3: pair bias = layernorm(z) . w_z  -> pb[h][q][k]. One block (128 thr) per (q,k).
__global__ void pair_bias_kernel(const float* __restrict__ z, const float* __restrict__ wz,
                                 const float* __restrict__ lnw, const float* __restrict__ lnb,
                                 float* __restrict__ pb) {
    int pid = blockIdx.x;                 // q*256 + kk
    int q = pid >> 8, kk = pid & 255;
    int t = threadIdx.x;                  // 0..127
    float x = z[pid * 128 + t];
    __shared__ float rs[128], rq[128], zn[128];
    rs[t] = x; rq[t] = x * x;
    __syncthreads();
    for (int st = 64; st > 0; st >>= 1) {
        if (t < st) { rs[t] += rs[t + st]; rq[t] += rq[t + st]; }
        __syncthreads();
    }
    float mean = rs[0] * (1.0f / 128.0f);
    float var  = rq[0] * (1.0f / 128.0f) - mean * mean;
    float rstd = rsqrtf(var + 1e-5f);
    zn[t] = (x - mean) * rstd * lnw[t] + lnb[t];
    __syncthreads();
    if (t < HEADS) {
        float acc = 0.0f;
        #pragma unroll 8
        for (int c = 0; c < 128; ++c) acc += zn[c] * wz[c * HEADS + t];
        pb[(t * 256 + q) * 256 + kk] = acc;
    }
}

// ---------------------------------------------------------------------------
// Kernel 4: fused Q/K/V/G projections. One wave per 16x16 output tile, K=256.
//  weight 0 -> q f16 [m][n]            (scale prefolded)
//  weight 1 -> k f16 [m][n]            (row-major == BT layout for QK^T)
//  weight 2 -> v f16 transposed vT[s][h][d][r]  (BT layout for attn*V)
//  weight 3 -> gate pre-act f16 [m][n] with b_g folded
__global__ void __launch_bounds__(256) proj_kernel(
    const _Float16* __restrict__ sn, const _Float16* __restrict__ wT,
    const float* __restrict__ bg,
    _Float16* __restrict__ qb, _Float16* __restrict__ kb,
    _Float16* __restrict__ vT, _Float16* __restrict__ gp) {
    int wid  = blockIdx.x * 8 + (threadIdx.x >> 5);    // 131072 waves
    int lane = threadIdx.x & 31;
    int weight = wid >> 15;
    int tile   = wid & 32767;
    int mT = tile >> 4, nT = tile & 15;
    int m0 = mT * 16, n0 = nT * 16;

    const _Float16* aRow = sn + (m0 + (lane & 15)) * 256;
    const _Float16* bRow = wT + weight * 65536 + (n0 + (lane & 15)) * 256;
    v8f acc = {0.f,0.f,0.f,0.f,0.f,0.f,0.f,0.f};
    #pragma unroll
    for (int k = 0; k < 256; k += 32) {
        v16h a = load_frag_A(aRow, k, lane);
        v16h b = load_frag_B(bRow, k, lane);
        acc = wmma_f16(a, b, acc);
    }
    int n = n0 + (lane & 15);
    int mAdd = (lane >= 16) ? 8 : 0;
    #pragma unroll
    for (int i = 0; i < 8; ++i) {
        int m = m0 + i + mAdd;
        float val = acc[i];
        if (weight == 0) {
            qb[m * 256 + n] = (_Float16)val;
        } else if (weight == 1) {
            kb[m * 256 + n] = (_Float16)val;
        } else if (weight == 2) {
            int si = m >> 8, r = m & 255, h = n >> 5, d = n & 31;
            vT[((si * HEADS + h) * DH + d) * 256 + r] = (_Float16)val;
        } else {
            gp[m * 256 + n] = (_Float16)(val + bg[n]);
        }
    }
}

// ---------------------------------------------------------------------------
// Kernel 5: attention. One wave per (s, h, 16 q-rows).
__global__ void __launch_bounds__(32) attn_kernel(
    const _Float16* __restrict__ qb, const _Float16* __restrict__ kb,
    const _Float16* __restrict__ vT, const _Float16* __restrict__ gp,
    const float* __restrict__ mask, const float* __restrict__ pb,
    _Float16* __restrict__ og) {
    int qT = blockIdx.x & 15;
    int h  = (blockIdx.x >> 4) & 7;
    int si = blockIdx.x >> 7;
    int lane = threadIdx.x;
    int laneN = lane & 15;
    int mAdd = (lane >= 16) ? 8 : 0;
    int q0 = qT * 16;

    __shared__ _Float16 s_attn[16 * 256];   // attn probs tile, 8KB

    // Q fragment: A 16x32, K = head dim (single WMMA K-step)
    const _Float16* qRow = qb + ((si * R_DIM + q0 + laneN) * 256 + h * DH);
    v16h aQ = load_frag_A(qRow, 0, lane);

    // logits: 16 tiles of 16x16 along k
    v8f acc[16];
    const v8f vzero = {0.f,0.f,0.f,0.f,0.f,0.f,0.f,0.f};
    #pragma unroll
    for (int t = 0; t < 16; ++t) {
        const _Float16* kRow = kb + ((si * R_DIM + t * 16 + laneN) * 256 + h * DH);
        v16h bK = load_frag_B(kRow, 0, lane);
        acc[t] = wmma_f16(aQ, bK, vzero);
    }
    // add mask bias + pair bias
    #pragma unroll
    for (int t = 0; t < 16; ++t) {
        int kk = t * 16 + laneN;
        float mb = INFV * (mask[si * R_DIM + kk] - 1.0f);
        #pragma unroll
        for (int i = 0; i < 8; ++i) {
            int qi = q0 + i + mAdd;
            acc[t][i] += mb + pb[(h * 256 + qi) * 256 + kk];
        }
    }
    // softmax over k (row = fixed vgpr i within a 16-lane group)
    float rinv[8];
    #pragma unroll
    for (int i = 0; i < 8; ++i) {
        float mx = -3.0e38f;
        #pragma unroll
        for (int t = 0; t < 16; ++t) mx = fmaxf(mx, acc[t][i]);
        mx = fmaxf(mx, __shfl_xor(mx, 1, 32));
        mx = fmaxf(mx, __shfl_xor(mx, 2, 32));
        mx = fmaxf(mx, __shfl_xor(mx, 4, 32));
        mx = fmaxf(mx, __shfl_xor(mx, 8, 32));
        float sm = 0.0f;
        #pragma unroll
        for (int t = 0; t < 16; ++t) {
            float e = __expf(acc[t][i] - mx);
            acc[t][i] = e;
            sm += e;
        }
        sm += __shfl_xor(sm, 1, 32);
        sm += __shfl_xor(sm, 2, 32);
        sm += __shfl_xor(sm, 4, 32);
        sm += __shfl_xor(sm, 8, 32);
        rinv[i] = 1.0f / sm;
    }
    // write normalized probs to LDS as f16 [qLocal][kk]
    #pragma unroll
    for (int t = 0; t < 16; ++t) {
        #pragma unroll
        for (int i = 0; i < 8; ++i) {
            s_attn[(i + mAdd) * 256 + t * 16 + laneN] = (_Float16)(acc[t][i] * rinv[i]);
        }
    }
    __syncthreads();

    // O = attn x V : A 16x256 (from LDS), B = vT rows (d-major, k contiguous)
    const _Float16* aRow = s_attn + laneN * 256;
    #pragma unroll
    for (int nt = 0; nt < 2; ++nt) {
        const _Float16* vRow = vT + ((si * HEADS + h) * DH + nt * 16 + laneN) * 256;
        v8f oacc = {0.f,0.f,0.f,0.f,0.f,0.f,0.f,0.f};
        #pragma unroll
        for (int k = 0; k < 256; k += 32) {
            v16h a = load_frag_A(aRow, k, lane);   // ds_load_b128 from LDS
            v16h b = load_frag_B(vRow, k, lane);
            oacc = wmma_f16(a, b, oacc);
        }
        int d = nt * 16 + laneN;
        #pragma unroll
        for (int i = 0; i < 8; ++i) {
            int qi = q0 + i + mAdd;
            int tok = si * R_DIM + qi;
            int x = h * DH + d;
            float gpre = (float)gp[tok * 256 + x];
            float gate = 1.0f / (1.0f + __expf(-gpre));
            og[tok * 256 + x] = (_Float16)(oacc[i] * gate);
        }
    }
}

// ---------------------------------------------------------------------------
// Kernel 6: output projection: out = og[32768,256] * w_o + b_o  (fp32 store)
__global__ void __launch_bounds__(256) out_gemm_kernel(
    const _Float16* __restrict__ og, const _Float16* __restrict__ woT,
    const float* __restrict__ bo, float* __restrict__ out) {
    int wid  = blockIdx.x * 8 + (threadIdx.x >> 5);    // 32768 waves
    int lane = threadIdx.x & 31;
    int mT = wid >> 4, nT = wid & 15;
    int m0 = mT * 16, n0 = nT * 16;
    const _Float16* aRow = og + (m0 + (lane & 15)) * 256;
    const _Float16* bRow = woT + (n0 + (lane & 15)) * 256;
    v8f acc = {0.f,0.f,0.f,0.f,0.f,0.f,0.f,0.f};
    #pragma unroll
    for (int k = 0; k < 256; k += 32) {
        v16h a = load_frag_A(aRow, k, lane);
        v16h b = load_frag_B(bRow, k, lane);
        acc = wmma_f16(a, b, acc);
    }
    int n = n0 + (lane & 15);
    int mAdd = (lane >= 16) ? 8 : 0;
    float bias = bo[n];
    #pragma unroll
    for (int i = 0; i < 8; ++i) {
        out[(m0 + i + mAdd) * 256 + n] = acc[i] + bias;
    }
}

// ---------------------------------------------------------------------------
extern "C" void kernel_launch(void* const* d_in, const int* in_sizes, int n_in,
                              void* d_out, int out_size, void* d_ws, size_t ws_size,
                              hipStream_t stream) {
    (void)in_sizes; (void)n_in; (void)out_size; (void)ws_size;
    const float* s_in  = (const float*)d_in[0];
    const float* z_in  = (const float*)d_in[1];
    const float* mask  = (const float*)d_in[2];
    const float* lnsw  = (const float*)d_in[3];
    const float* lnsb  = (const float*)d_in[4];
    const float* lnzw  = (const float*)d_in[5];
    const float* lnzb  = (const float*)d_in[6];
    const float* w_z   = (const float*)d_in[7];
    const float* w_q   = (const float*)d_in[8];
    const float* w_k   = (const float*)d_in[9];
    const float* w_v   = (const float*)d_in[10];
    const float* w_g   = (const float*)d_in[11];
    const float* b_g   = (const float*)d_in[12];
    const float* w_o   = (const float*)d_in[13];
    const float* b_o   = (const float*)d_in[14];
    float* out = (float*)d_out;

    // workspace layout (bytes)
    char* ws = (char*)d_ws;
    const size_t SZ_ACT = (size_t)NTOK * 256 * 2;      // 16 MB per f16 activation buf
    size_t off = 0;
    _Float16* wT  = (_Float16*)(ws + off); off += 5 * 65536 * 2;
    off = (off + 1023) & ~(size_t)1023;
    _Float16* sn  = (_Float16*)(ws + off); off += SZ_ACT;
    _Float16* qb  = (_Float16*)(ws + off); off += SZ_ACT;
    _Float16* kb  = (_Float16*)(ws + off); off += SZ_ACT;
    _Float16* vT  = (_Float16*)(ws + off); off += SZ_ACT;
    _Float16* gp  = (_Float16*)(ws + off); off += SZ_ACT;
    _Float16* og  = (_Float16*)(ws + off); off += SZ_ACT;
    float*    pb  = (float*)(ws + off);    off += (size_t)HEADS * 256 * 256 * 4;

    prep_w_kernel<<<1280, 256, 0, stream>>>(w_q, w_k, w_v, w_g, w_o, wT);
    ln_s_kernel<<<NTOK, 256, 0, stream>>>(s_in, lnsw, lnsb, sn);
    pair_bias_kernel<<<65536, 128, 0, stream>>>(z_in, w_z, lnzw, lnzb, pb);
    proj_kernel<<<16384, 256, 0, stream>>>(sn, wT, b_g, qb, kb, vT, gp);
    attn_kernel<<<S_DIM * HEADS * 16, 32, 0, stream>>>(qb, kb, vT, gp, mask, pb, og);
    out_gemm_kernel<<<4096, 256, 0, stream>>>(og, wT + 4 * 65536, b_o, out);
}
